// CrossAttentionBlock_85349590106342
// MI455X (gfx1250) — compile-verified, matrix operand order checked by reading, not run on hardware
//
#include <hip/hip_runtime.h>
#include <hip/hip_bf16.h>

typedef __attribute__((ext_vector_type(16))) _Float16 v16h;
typedef __attribute__((ext_vector_type(8)))  _Float16 v8h;
typedef __attribute__((ext_vector_type(8)))  float    v8f;
typedef __attribute__((ext_vector_type(4)))  float    v4f;

union V16 { v16h v; v8h p[2]; };

#define WMMA_F16(A, B, C) \
    __builtin_amdgcn_wmma_f32_16x16x32_f16(false, (A), false, (B), (short)0, (C), false, false)

// ---------------------------------------------------------------------------
#define DIM    256
#define HEADS  8
#define HD     32
#define NSEQ   2048
#define SCALE  0.0625f     // 256^-0.5, full-dim scale per reference

// CDNA5 async global->LDS staging (ASYNCcnt-tracked, no VGPR round trip).
#define USE_ASYNC_LDS 1

__device__ __forceinline__ void cp16_lds(_Float16* lds, const _Float16* g) {
#if USE_ASYNC_LDS
    unsigned ldsa = (unsigned)(unsigned long long)lds;   // flat addr low 32b = LDS offset
    asm volatile("global_load_async_to_lds_b128 %0, %1, off"
                 :: "v"(ldsa), "v"(g) : "memory");
#else
    *(v8h*)lds = *(const v8h*)g;
#endif
}
__device__ __forceinline__ void cp_wait_publish() {
#if USE_ASYNC_LDS
    asm volatile("s_wait_asynccnt 0x0" ::: "memory");
#endif
    __syncthreads();
}

__device__ __forceinline__ float wave_sum(float v) {
    #pragma unroll
    for (int o = 16; o > 0; o >>= 1) v += __shfl_xor(v, o);
    return v;
}

// ---------------------------------------------------------------------------
// fp32 -> fp16 weight conversion (Wq, Wk, Wv, Wo : 256x256 each)
// ---------------------------------------------------------------------------
__global__ __launch_bounds__(256) void cvt_w_kernel(
    const float* __restrict__ wq, const float* __restrict__ wk,
    const float* __restrict__ wv, const float* __restrict__ wo,
    _Float16* __restrict__ o)
{
    int i = blockIdx.x * 256 + threadIdx.x;            // grid 256 -> i < 65536
    o[i]          = (_Float16)wq[i];
    o[65536 + i]  = (_Float16)wk[i];
    o[131072 + i] = (_Float16)wv[i];
    o[196608 + i] = (_Float16)wo[i];
}

// ---------------------------------------------------------------------------
// LayerNorm over dim=256, one wave per token, fp16 output
// ---------------------------------------------------------------------------
__global__ __launch_bounds__(256) void ln_kernel(
    const float* __restrict__ x, const float* __restrict__ g,
    const float* __restrict__ bt, _Float16* __restrict__ xn)
{
    int lane = threadIdx.x & 31, wid = threadIdx.x >> 5;
    int tok = blockIdx.x * 8 + wid;                    // grid 6144 -> tok < 49152
    const float* xp = x + (size_t)tok * DIM + lane * 8;
    v4f a0 = *(const v4f*)xp;
    v4f a1 = *(const v4f*)(xp + 4);
    float s = 0.f;
    #pragma unroll
    for (int i = 0; i < 4; i++) s += a0[i] + a1[i];
    s = wave_sum(s);
    float mu = s * (1.f / 256.f);
    float vs = 0.f;
    #pragma unroll
    for (int i = 0; i < 4; i++) { float d0 = a0[i]-mu, d1 = a1[i]-mu; vs += d0*d0 + d1*d1; }
    vs = wave_sum(vs);
    float rstd = rsqrtf(vs * (1.f / 256.f) + 1e-5f);
    v4f g0 = *(const v4f*)(g  + lane*8), g1 = *(const v4f*)(g  + lane*8 + 4);
    v4f b0 = *(const v4f*)(bt + lane*8), b1 = *(const v4f*)(bt + lane*8 + 4);
    v8h o;
    #pragma unroll
    for (int i = 0; i < 4; i++) {
        o[i]   = (_Float16)((a0[i]-mu)*rstd*g0[i] + b0[i]);
        o[4+i] = (_Float16)((a1[i]-mu)*rstd*g1[i] + b1[i]);
    }
    *(v8h*)(xn + (size_t)tok * DIM + lane * 8) = o;
}

// ---------------------------------------------------------------------------
// WMMA GEMM: C[M x 256] = A(f16) @ W^T(f16 256x256) + bias.
// Block = 8 waves, tile 128(M) x 64(N); the 64-row W slab (shared by all 8
// waves) is staged to LDS once (padded stride 264 halves -> conflict-free).
// mode 0: out16 row-major [m][n]            (Q / K projections)
// mode 1: out16 transposed  [b][h][d][key]  (V projection -> V^T)
// mode 2: out32 row-major [m][n]            (final Wo projection -> d_out)
// ---------------------------------------------------------------------------
#define WLP 264                                        // W LDS row stride (halves)

__global__ __launch_bounds__(256) void gemm_kernel(
    const _Float16* __restrict__ A, const _Float16* __restrict__ W,
    const float* __restrict__ bias, float bias_scale,
    _Float16* __restrict__ out16, float* __restrict__ out32,
    int mode, int stream_off)
{
    __shared__ _Float16 wlds[64 * WLP];
    int t = threadIdx.x;
    int lane = t & 31, wid = t >> 5;
    int half = lane >> 4, ln = lane & 15;
    int m0 = (blockIdx.x * 8 + wid) * 16;
    int n0 = blockIdx.y * 64;

    // Stage 64 x 256 W slab into LDS (2048 16-byte chunks, 8 per thread)
    #pragma unroll
    for (int j = 0; j < 8; j++) {
        int cid = j * 256 + t;
        int row = cid >> 5, grp = (cid & 31) * 8;
        cp16_lds(&wlds[row * WLP + grp], W + (size_t)(n0 + row) * DIM + grp);
    }
    cp_wait_publish();

    int mrow = m0 + ln;
    int arow = (mode == 2) ? mrow
                           : ((mrow >> 11) * 6144 + stream_off + (mrow & 2047));
    const _Float16* ap = A + (size_t)arow * DIM + half * 8;

    v8f acc[4] = {v8f{}, v8f{}, v8f{}, v8f{}};
    for (int k0 = 0; k0 < DIM; k0 += 32) {
        V16 af;
        af.p[0] = *(const v8h*)(ap + k0);
        af.p[1] = *(const v8h*)(ap + k0 + 16);
        #pragma unroll
        for (int nt = 0; nt < 4; nt++) {
            const _Float16* wp = wlds + (nt*16 + ln) * WLP + k0 + half * 8;
            V16 wf;
            wf.p[0] = *(const v8h*)wp;
            wf.p[1] = *(const v8h*)(wp + 16);
            acc[nt] = WMMA_F16(af.v, wf.v, acc[nt]);
        }
    }

    #pragma unroll
    for (int nt = 0; nt < 4; nt++) {
        int n = n0 + nt*16 + ln;
        float bv = bias[n] * bias_scale;
        if (mode == 0) {
            #pragma unroll
            for (int r = 0; r < 8; r++) {
                int m = m0 + half*8 + r;
                out16[(size_t)m * DIM + n] = (_Float16)(acc[nt][r] + bv);
            }
        } else if (mode == 1) {
            v8h pk;
            #pragma unroll
            for (int r = 0; r < 8; r++) pk[r] = (_Float16)(acc[nt][r] + bv);
            int mbase = m0 + half*8;
            int bb = mbase >> 11, pos = mbase & 2047;
            int hh = n >> 5, dd = n & 31;
            *(v8h*)(out16 + ((size_t)((bb*HEADS + hh)*HD + dd)) * NSEQ + pos) = pk;
        } else {
            #pragma unroll
            for (int r = 0; r < 8; r++) {
                int m = m0 + half*8 + r;
                out32[(size_t)m * DIM + n] = acc[nt][r] + bv;
            }
        }
    }
}

// ---------------------------------------------------------------------------
// Transposed flash attention with LDS-shared K/V^T.
// Block = one (b,h) x 128 queries (8 waves x 16-query tiles). 64-key K and
// V^T chunks are async-staged to LDS (double-buffered) and consumed by all
// 8 waves -> 8x less K/V read traffic than per-wave global loads.
// S^T = K_chunk @ Q^T keeps softmax per-lane (+1 xor-16 shuffle) and the
// exp'd C-layout tile IS the B-fragment of O^T = V^T @ P^T (no shuffles).
// ---------------------------------------------------------------------------
#define KP     40                       // K tile row stride (halves), 64 rows
#define VP     72                       // V^T tile row stride (halves), 32 rows
#define KT_SZ  (64 * KP)                // 2560
#define VT_SZ  (32 * VP)                // 2304
#define BUF_SZ (KT_SZ + VT_SZ)          // 4864 halves per buffer

__global__ __launch_bounds__(256) void attn_kernel(
    const _Float16* __restrict__ Q,
    const _Float16* __restrict__ K1, const _Float16* __restrict__ Vt1,
    const _Float16* __restrict__ K2, const _Float16* __restrict__ Vt2,
    _Float16* __restrict__ Osum)
{
    __shared__ _Float16 smem[2 * BUF_SZ];
    int t = threadIdx.x;
    int lane = t & 31, wid = t >> 5;
    int half = lane >> 4, ln = lane & 15;
    int qblk = blockIdx.x & 15;                 // grid 1024 = b(8) x h(8) x 16
    int h    = (blockIdx.x >> 4) & 7;
    int b    = blockIdx.x >> 7;
    int q0   = qblk * 128 + wid * 16;

    // Q B-fragment (held for the whole kernel)
    const _Float16* qp = Q + (size_t)(b*NSEQ + q0 + ln) * DIM + h*HD + half*8;
    V16 qf;
    qf.p[0] = *(const v8h*)qp;
    qf.p[1] = *(const v8h*)(qp + 16);

    // chunk c in [0,64): pass = c>>5, keys (c&31)*64 .. +64
    auto stage = [&](int c, int buf) {
        int pass = c >> 5;
        int k0 = (c & 31) * 64;
        const _Float16* K  = pass ? K2  : K1;
        const _Float16* Vt = pass ? Vt2 : Vt1;
        _Float16* dstb = smem + buf * BUF_SZ;
        {   // K tile: 64 keys x 32 d  (one b128 per thread)
            int row = t >> 2, cg = (t & 3) * 8;
            cp16_lds(dstb + row*KP + cg,
                     K + (size_t)(b*NSEQ + k0 + row) * DIM + h*HD + cg);
        }
        {   // V^T tile: 32 d x 64 keys (one b128 per thread)
            int d = t >> 3, kg = (t & 7) * 8;
            cp16_lds(dstb + KT_SZ + d*VP + kg,
                     Vt + ((size_t)(b*HEADS + h)*HD + d) * NSEQ + k0 + kg);
        }
    };

    float mrun = -3.0e38f, lrun = 0.f;
    v8f o0 = {}, o1 = {}, out0 = {}, out1 = {};

    stage(0, 0);
    for (int c = 0; c < 64; c++) {
        if (c == 32) {                          // finalize pass 0, reset state
            float inv = 1.f / lrun;
            #pragma unroll
            for (int r = 0; r < 8; r++) { out0[r] += o0[r]*inv; out1[r] += o1[r]*inv; }
            o0 = v8f{}; o1 = v8f{};
            mrun = -3.0e38f; lrun = 0.f;
        }
        cp_wait_publish();                      // chunk c visible to all waves
        if (c + 1 < 64) stage(c + 1, (c + 1) & 1);

        const _Float16* kt = smem + (c & 1) * BUF_SZ;
        const _Float16* vt = kt + KT_SZ;
        #pragma unroll
        for (int s = 0; s < 64; s += 32) {
            // ---- S^T (32 keys x 16 queries) ----
            const _Float16* kq = kt + (size_t)(s + ln) * KP + half*8;
            V16 ka, kb;
            ka.p[0] = *(const v8h*)kq;
            ka.p[1] = *(const v8h*)(kq + 16);
            kb.p[0] = *(const v8h*)(kq + 16*KP);
            kb.p[1] = *(const v8h*)(kq + 16*KP + 16);
            v8f s0 = {}, s1 = {};
            s0 = WMMA_F16(ka.v, qf.v, s0);
            s1 = WMMA_F16(kb.v, qf.v, s1);

            // ---- online softmax over 32-key chunk ----
            float mx = -3.0e38f;
            #pragma unroll
            for (int r = 0; r < 8; r++) {
                s0[r] *= SCALE; s1[r] *= SCALE;
                mx = fmaxf(mx, fmaxf(s0[r], s1[r]));
            }
            mx = fmaxf(mx, __shfl_xor(mx, 16));
            float mnew = fmaxf(mrun, mx);
            float al = __expf(mrun - mnew);
            V16 pf;
            float ls = 0.f;
            #pragma unroll
            for (int r = 0; r < 8; r++) {
                float e0 = __expf(s0[r] - mnew);
                float e1 = __expf(s1[r] - mnew);
                ls += e0 + e1;
                pf.p[0][r] = (_Float16)e0;
                pf.p[1][r] = (_Float16)e1;
            }
            ls += __shfl_xor(ls, 16);
            lrun = lrun * al + ls;
            mrun = mnew;
            #pragma unroll
            for (int r = 0; r < 8; r++) { o0[r] *= al; o1[r] *= al; }

            // ---- O^T += V^T @ P^T ----
            const _Float16* vq = vt + (size_t)ln * VP + s + half*8;
            V16 va, vb;
            va.p[0] = *(const v8h*)vq;
            va.p[1] = *(const v8h*)(vq + 16);
            vb.p[0] = *(const v8h*)(vq + 16*VP);
            vb.p[1] = *(const v8h*)(vq + 16*VP + 16);
            o0 = WMMA_F16(va.v, pf.v, o0);
            o1 = WMMA_F16(vb.v, pf.v, o1);
        }
    }
    {   // finalize pass 1
        float inv = 1.f / lrun;
        #pragma unroll
        for (int r = 0; r < 8; r++) { out0[r] += o0[r]*inv; out1[r] += o1[r]*inv; }
    }

    // Store O1+O2, token-major f16 (contiguous along d -> two b128 stores)
    size_t tokbase = (size_t)(b*NSEQ + q0 + ln) * DIM + h*HD;
    v8h pk0, pk1;
    #pragma unroll
    for (int r = 0; r < 8; r++) {
        pk0[r] = (_Float16)out0[r];
        pk1[r] = (_Float16)out1[r];
    }
    *(v8h*)(Osum + tokbase + half*8)      = pk0;
    *(v8h*)(Osum + tokbase + 16 + half*8) = pk1;
}

// ---------------------------------------------------------------------------
extern "C" void kernel_launch(void* const* d_in, const int* in_sizes, int n_in,
                              void* d_out, int out_size, void* d_ws, size_t ws_size,
                              hipStream_t stream) {
    (void)in_sizes; (void)n_in; (void)out_size; (void)ws_size;

    const float* x   = (const float*)d_in[0];
    const float* lng = (const float*)d_in[1];
    const float* lnb = (const float*)d_in[2];
    const float* Wq  = (const float*)d_in[3];
    const float* bq  = (const float*)d_in[4];
    const float* Wk  = (const float*)d_in[5];
    const float* bk  = (const float*)d_in[6];
    const float* Wv  = (const float*)d_in[7];
    const float* bv  = (const float*)d_in[8];
    const float* Wo  = (const float*)d_in[9];
    const float* bo  = (const float*)d_in[10];
    float* out = (float*)d_out;

    // Workspace layout (f16 elements); total ~76 MB
    _Float16* ws   = (_Float16*)d_ws;
    _Float16* xn   = ws;                    // 49152*256 = 12,582,912
    _Float16* Qh   = xn   + 12582912;       // 16384*256 =  4,194,304 each below
    _Float16* K1h  = Qh   + 4194304;
    _Float16* Vt1  = K1h  + 4194304;
    _Float16* K2h  = Vt1  + 4194304;
    _Float16* Vt2  = K2h  + 4194304;
    _Float16* Osum = Vt2  + 4194304;
    _Float16* W16  = Osum + 4194304;        // 4 * 65536
    _Float16* Wq16 = W16;
    _Float16* Wk16 = W16 + 65536;
    _Float16* Wv16 = W16 + 131072;
    _Float16* Wo16 = W16 + 196608;

    cvt_w_kernel<<<256, 256, 0, stream>>>(Wq, Wk, Wv, Wo, W16);
    ln_kernel<<<6144, 256, 0, stream>>>(x, lng, lnb, xn);

    dim3 g(128, 4);
    // Q from origin stream; K/V from emap (pass 1) and lbp (pass 2)
    gemm_kernel<<<g, 256, 0, stream>>>(xn, Wq16, bq, 1.f, Qh,  nullptr, 0, 0);
    gemm_kernel<<<g, 256, 0, stream>>>(xn, Wk16, bk, 1.f, K1h, nullptr, 0, 2048);
    gemm_kernel<<<g, 256, 0, stream>>>(xn, Wv16, bv, 1.f, Vt1, nullptr, 1, 2048);
    gemm_kernel<<<g, 256, 0, stream>>>(xn, Wk16, bk, 1.f, K2h, nullptr, 0, 4096);
    gemm_kernel<<<g, 256, 0, stream>>>(xn, Wv16, bv, 1.f, Vt2, nullptr, 1, 4096);

    attn_kernel<<<1024, 256, 0, stream>>>(Qh, K1h, Vt1, K2h, Vt2, Osum);

    // out = (O1+O2) @ Wo^T + 2*bo   (linearity of the two passes' output proj)
    gemm_kernel<<<g, 256, 0, stream>>>(Osum, Wo16, bo, 2.f, nullptr, out, 2, 0);
}
